// EdgeClassifier_59983513256041
// MI455X (gfx1250) — matrix operand order repaired and checked by reading, not compiled, not run on hardware
//
#include <hip/hip_runtime.h>

#define NN 100000     // nodes
#define NE 1000000    // edges
#define D  128        // D_IN == D_H
#define NC 16         // classes

typedef __attribute__((ext_vector_type(16))) __bf16 v16bf;
typedef __attribute__((ext_vector_type(8)))  __bf16 v8bf;
typedef __attribute__((ext_vector_type(4)))  __bf16 v4bf;
typedef __attribute__((ext_vector_type(8)))  float  v8f;

// ---- WMMA helpers -------------------------------------------------------
__device__ __forceinline__ v8f wmma_bf16(v16bf a, v16bf b, v8f c) {
  return __builtin_amdgcn_wmma_f32_16x16x32_bf16(
      /*neg_a=*/false, a, /*neg_b=*/false, b,
      /*c_mod=*/(short)0, c, /*reuse_a=*/false, /*reuse_b=*/false);
}

// A fragment (16x32 bf16): lane<16 holds row m=lane, K {k0..k0+7, k0+16..k0+23};
// lane>=16 holds row m=lane-16, K {k0+8..k0+15, k0+24..k0+31}.
__device__ __forceinline__ v16bf load_a_frag(const __bf16* row, int k0, int hi) {
  const int base = k0 + (hi << 3);
  v8bf lo = *(const v8bf*)(row + base);
  v8bf h2 = *(const v8bf*)(row + base + 16);
  return __builtin_shufflevector(lo, h2, 0,1,2,3,4,5,6,7,8,9,10,11,12,13,14,15);
}

// B fragment (32x16 bf16): lane<16 = col n=lane, K {k0..k0+7, k0+16..k0+23};
// lane>=16 = K {k0+8..k0+15, k0+24..k0+31}.  row = weight row [out][in] row-major.
__device__ __forceinline__ v16bf load_b_frag(const __bf16* row, int k0, int hi) {
  const int base = k0 + (hi << 3);
  v8bf lo = *(const v8bf*)(row + base);
  v8bf h2 = *(const v8bf*)(row + base + 16);
  return __builtin_shufflevector(lo, h2, 0,1,2,3,4,5,6,7,8,9,10,11,12,13,14,15);
}

// native f32 atomic add (global_atomic_add_f32); avoid CAS-loop lowering
__device__ __forceinline__ void fadd_atomic(float* p, float v) {
  unsafeAtomicAdd(p, v);
}

// ---- utility kernels ----------------------------------------------------
__global__ void k_zero_f32(float* __restrict__ p, int n) {
  int i = blockIdx.x * blockDim.x + threadIdx.x;
  if (i < n) p[i] = 0.0f;
}

__global__ void k_f32_to_bf16(const float* __restrict__ in, __bf16* __restrict__ out, int n) {
  int i = blockIdx.x * blockDim.x + threadIdx.x;
  if (i < n) out[i] = (__bf16)in[i];
}

// scatter: 32 lanes per edge, 4 feats per lane, f32 atomics into agg
__global__ void k_scatter(const __bf16* __restrict__ x,
                          const long long* __restrict__ src,
                          const long long* __restrict__ dst,
                          float* __restrict__ agg, float* __restrict__ cnt) {
  long long t = (long long)blockIdx.x * blockDim.x + threadIdx.x;
  int e = (int)(t >> 5);
  int l = (int)(t & 31);
  if (e >= NE) return;
  long long s = src[e], d = dst[e];
  const __bf16* xr = x + (size_t)s * D + l * 4;
  float* ar = agg + (size_t)d * D + l * 4;
  v4bf v = *(const v4bf*)xr;
#pragma unroll
  for (int i = 0; i < 4; ++i) fadd_atomic(ar + i, (float)v[i]);
  if (l == 0) fadd_atomic(cnt + d, 1.0f);
}

__global__ void k_mean(const float* __restrict__ agg, const float* __restrict__ cnt,
                       __bf16* __restrict__ out) {
  int i = blockIdx.x * blockDim.x + threadIdx.x;
  if (i >= NN * D) return;
  int node = i >> 7;
  float inv = 1.0f / fmaxf(cnt[node], 1.0f);
  out[i] = (__bf16)(agg[i] * inv);
}

// ---- fused SAGE layer: one wave per 16-node tile ------------------------
// h    = relu(mean @ Wl.T + bl + x @ Wr.T)          (kept in per-wave LDS)
// xout = relu(x @ Wc[:, :128].T + h @ Wc[:,128:].T + bc)   (bf16)
__global__ __launch_bounds__(256) void k_sage(const __bf16* __restrict__ xbf,
                                              const __bf16* __restrict__ meanbf,
                                              const __bf16* __restrict__ wl,
                                              const float*  __restrict__ bl,
                                              const __bf16* __restrict__ wr,
                                              const __bf16* __restrict__ wc,
                                              const float*  __restrict__ bc,
                                              __bf16* __restrict__ xout) {
  __shared__ __align__(16) __bf16 hst[8][16 * D];   // 4 KB per wave
  const int wave = threadIdx.x >> 5;
  const int lane = threadIdx.x & 31;
  const int tile = blockIdx.x * 8 + wave;
  if (tile >= NN / 16) return;                      // uniform per wave
  const int hi = lane >> 4;
  const int m  = lane & 15;
  const int node = tile * 16 + m;
  const __bf16* xrow = xbf    + (size_t)node * D;
  const __bf16* mrow = meanbf + (size_t)node * D;
  const __bf16* hrow = &hst[wave][m * D];

  // phase 1: h tile -> LDS (bf16, with relu)
#pragma unroll
  for (int nt = 0; nt < 8; ++nt) {
    v8f acc = {};
    const __bf16* blrow = wl + (size_t)(nt * 16 + m) * D;
    const __bf16* brrow = wr + (size_t)(nt * 16 + m) * D;
#pragma unroll
    for (int k0 = 0; k0 < D; k0 += 32) {
      acc = wmma_bf16(load_a_frag(mrow, k0, hi), load_b_frag(blrow, k0, hi), acc);
      acc = wmma_bf16(load_a_frag(xrow, k0, hi), load_b_frag(brrow, k0, hi), acc);
    }
    float bias = bl[nt * 16 + m];
#pragma unroll
    for (int r = 0; r < 8; ++r) {
      float v = fmaxf(acc[r] + bias, 0.0f);
      hst[wave][(r + hi * 8) * D + nt * 16 + m] = (__bf16)v;   // C layout scatter
    }
  }
  // DS ops are in-order within a wave; only this wave touches hst[wave].

  // phase 2: combine [x, h] @ Wc.T
#pragma unroll
  for (int nt = 0; nt < 8; ++nt) {
    v8f acc = {};
    const __bf16* wcrow = wc + (size_t)(nt * 16 + m) * (2 * D);
#pragma unroll
    for (int k0 = 0; k0 < D; k0 += 32) {
      acc = wmma_bf16(load_a_frag(xrow, k0, hi), load_b_frag(wcrow,     k0, hi), acc);
      acc = wmma_bf16(load_a_frag(hrow, k0, hi), load_b_frag(wcrow + D, k0, hi), acc);
    }
    float bias = bc[nt * 16 + m];
#pragma unroll
    for (int r = 0; r < 8; ++r) {
      float v = fmaxf(acc[r] + bias, 0.0f);
      xout[(size_t)(tile * 16 + r + hi * 8) * D + nt * 16 + m] = (__bf16)v;
    }
  }
}

// ---- edge head: one wave per 16-edge tile, K=256 ------------------------
__global__ __launch_bounds__(256) void k_edge(const __bf16* __restrict__ xbf,
                                              const long long* __restrict__ src,
                                              const long long* __restrict__ dst,
                                              const __bf16* __restrict__ wo,
                                              const float*  __restrict__ bo,
                                              float* __restrict__ out) {
  const int wave = threadIdx.x >> 5;
  const int lane = threadIdx.x & 31;
  const int tile = blockIdx.x * 8 + wave;
  if (tile >= NE / 16) return;
  const int hi = lane >> 4;
  const int m  = lane & 15;
  const int e  = tile * 16 + m;
  long long s = src[e], d = dst[e];
  const __bf16* srow = xbf + (size_t)s * D;
  const __bf16* drow = xbf + (size_t)d * D;
  const __bf16* wrow = wo + (size_t)m * (2 * D);   // column n = m of W_o.T
  v8f acc = {};
#pragma unroll
  for (int k0 = 0; k0 < D; k0 += 32)
    acc = wmma_bf16(load_a_frag(srow, k0, hi), load_b_frag(wrow, k0, hi), acc);
#pragma unroll
  for (int k0 = 0; k0 < D; k0 += 32)
    acc = wmma_bf16(load_a_frag(drow, k0, hi), load_b_frag(wrow, D + k0, hi), acc);
  float bias = bo[m];
#pragma unroll
  for (int r = 0; r < 8; ++r)
    out[(size_t)(tile * 16 + r + hi * 8) * NC + m] = acc[r] + bias;
}

// ---- host ---------------------------------------------------------------
extern "C" void kernel_launch(void* const* d_in, const int* in_sizes, int n_in,
                              void* d_out, int out_size, void* d_ws, size_t ws_size,
                              hipStream_t stream) {
  const float*     x   = (const float*)d_in[0];
  const long long* ei  = (const long long*)d_in[1];
  const float*     Wl0 = (const float*)d_in[2];
  const float*     bl0 = (const float*)d_in[3];
  const float*     Wr0 = (const float*)d_in[4];
  const float*     Wl1 = (const float*)d_in[5];
  const float*     bl1 = (const float*)d_in[6];
  const float*     Wr1 = (const float*)d_in[7];
  const float*     Wc  = (const float*)d_in[8];
  const float*     bc  = (const float*)d_in[9];
  const float*     Wo  = (const float*)d_in[10];
  const float*     bo  = (const float*)d_in[11];
  const long long* srcI = ei;
  const long long* dstI = ei + NE;

  char*  w   = (char*)d_ws;
  size_t off = 0;
  auto alloc = [&](size_t bytes) -> void* {
    void* p = w + off;
    off = (off + bytes + 255) & ~(size_t)255;
    return p;
  };
  __bf16* xbfA   = (__bf16*)alloc((size_t)NN * D * 2);
  __bf16* xbfB   = (__bf16*)alloc((size_t)NN * D * 2);
  __bf16* meanbf = (__bf16*)alloc((size_t)NN * D * 2);
  float*  agg    = (float*) alloc((size_t)NN * D * 4);
  float*  cnt    = (float*) alloc((size_t)NN * 4);
  __bf16* wl0    = (__bf16*)alloc((size_t)D * D * 2);
  __bf16* wr0    = (__bf16*)alloc((size_t)D * D * 2);
  __bf16* wl1    = (__bf16*)alloc((size_t)D * D * 2);
  __bf16* wr1    = (__bf16*)alloc((size_t)D * D * 2);
  __bf16* wcb    = (__bf16*)alloc((size_t)D * 2 * D * 2);
  __bf16* wob    = (__bf16*)alloc((size_t)NC * 2 * D * 2);

  const int TB = 256;
  auto cdiv = [](int a, int b) { return (a + b - 1) / b; };

  // bf16 conversions (inputs are never mutated; ws is fully rewritten each call)
  k_f32_to_bf16<<<cdiv(NN * D, TB), TB, 0, stream>>>(x,   xbfA, NN * D);
  k_f32_to_bf16<<<cdiv(D * D, TB),  TB, 0, stream>>>(Wl0, wl0,  D * D);
  k_f32_to_bf16<<<cdiv(D * D, TB),  TB, 0, stream>>>(Wr0, wr0,  D * D);
  k_f32_to_bf16<<<cdiv(D * D, TB),  TB, 0, stream>>>(Wl1, wl1,  D * D);
  k_f32_to_bf16<<<cdiv(D * D, TB),  TB, 0, stream>>>(Wr1, wr1,  D * D);
  k_f32_to_bf16<<<cdiv(D * 2 * D, TB), TB, 0, stream>>>(Wc, wcb, D * 2 * D);
  k_f32_to_bf16<<<cdiv(NC * 2 * D, TB), TB, 0, stream>>>(Wo, wob, NC * 2 * D);

  const int scatterBlocks = cdiv(NE * 32, TB);
  const int sageBlocks    = cdiv(NN / 16, 8);
  const int edgeBlocks    = cdiv(NE / 16, 8);

  // ---- layer 0 ----
  k_zero_f32<<<cdiv(NN * D, TB), TB, 0, stream>>>(agg, NN * D);
  k_zero_f32<<<cdiv(NN, TB),     TB, 0, stream>>>(cnt, NN);
  k_scatter<<<scatterBlocks, TB, 0, stream>>>(xbfA, srcI, dstI, agg, cnt);
  k_mean<<<cdiv(NN * D, TB), TB, 0, stream>>>(agg, cnt, meanbf);
  k_sage<<<sageBlocks, TB, 0, stream>>>(xbfA, meanbf, wl0, bl0, wr0, wcb, bc, xbfB);

  // ---- layer 1 ----
  k_zero_f32<<<cdiv(NN * D, TB), TB, 0, stream>>>(agg, NN * D);
  k_zero_f32<<<cdiv(NN, TB),     TB, 0, stream>>>(cnt, NN);
  k_scatter<<<scatterBlocks, TB, 0, stream>>>(xbfB, srcI, dstI, agg, cnt);
  k_mean<<<cdiv(NN * D, TB), TB, 0, stream>>>(agg, cnt, meanbf);
  k_sage<<<sageBlocks, TB, 0, stream>>>(xbfB, meanbf, wl1, bl1, wr1, wcb, bc, xbfA);

  // ---- edge head ----
  k_edge<<<edgeBlocks, TB, 0, stream>>>(xbfA, srcI, dstI, wob, bo, (float*)d_out);
}
